// CategoricalRSSM_18047452577978
// MI455X (gfx1250) — matrix operand order repaired and checked by reading, not compiled, not run on hardware
//
#include <hip/hip_runtime.h>
#include <stdint.h>

// CategoricalRSSM for MI455X (gfx1250): T=64 sequential steps, each step is
// 6 fused kernels. All GEMMs run on v_wmma_f32_16x16x32_bf16 (f32 accum).
// Weights converted to bf16 once per call (resident in 192MB L2 across steps).
// A-tiles (activations, shared by all 8 waves of a block) are staged into LDS
// with global_load_async_to_lds_b128 (ASYNCcnt) and consumed via ds_load_b128,
// so the VMEM pipe is dedicated to streaming weight fragments from L2.

#define Bb 1024
#define Hh 256
#define Ss 256
#define Ee 512
#define Tt 64

typedef __attribute__((ext_vector_type(16))) __bf16 v16bf;
typedef __attribute__((ext_vector_type(8)))  __bf16 v8bf;
typedef __attribute__((ext_vector_type(8)))  float  v8f;

static __device__ __forceinline__ __bf16 f2bf(float f) {
  uint32_t u = __builtin_bit_cast(uint32_t, f);
  uint32_t r = (u + 0x7FFFu + ((u >> 16) & 1u)) >> 16;
  return __builtin_bit_cast(__bf16, (uint16_t)r);
}

static __device__ __forceinline__ uint32_t hashu(uint32_t x) {
  x ^= x >> 17; x *= 0xED5AD4BBu;
  x ^= x >> 11; x *= 0xAC4C1B51u;
  x ^= x >> 15; x *= 0x31848BABu;
  x ^= x >> 14; return x;
}
static __device__ __forceinline__ float gumbel_from(uint32_t h) {
  float u = (float)(h >> 8) * (1.0f / 16777216.0f) + 1e-7f;
  return -__logf(-__logf(u));
}

// ---------------- async copy: 16 x KLEN bf16 activation tile -> LDS ---------
// 256 threads, 16 B (b128) per thread per pass; ASYNCcnt-tracked DMA-style
// copy (no VGPR staging). Caller must s_wait_asynccnt + barrier before use.
template <int KLEN>
static __device__ __forceinline__ void stage_tile(__bf16* sdst,
                                                  const __bf16* __restrict__ src,
                                                  int lda, int row0, int tid) {
  constexpr int TOT = KLEN * 16;          // elements in tile
  uint32_t ldsb = (uint32_t)(uintptr_t)sdst;  // low 32 bits = LDS byte address
#pragma unroll
  for (int p = 0; p < TOT / 2048; p++) {  // 2048 elems per 256-thread pass
    int e = p * 2048 + tid * 8;           // 8 bf16 = 16 B chunk
    int r = e / KLEN, c = e % KLEN;
    unsigned long long g =
        (unsigned long long)(uintptr_t)(src + (size_t)(row0 + r) * lda + c);
    asm volatile("global_load_async_to_lds_b128 %0, %1, off"
                 :: "v"(ldsb + (uint32_t)(e * 2)), "v"(g)
                 : "memory");
  }
}
static __device__ __forceinline__ void async_wait_barrier() {
  asm volatile("s_wait_asynccnt 0x0" ::: "memory");
  __syncthreads();
}

// A-fragment: 16x32 bf16 tile (row-major [.., lda]); works for LDS or global.
// ISA layout: lanes 0-15 -> M=lane, K = k0+0..7 (v0..3) and k0+16..23 (v4..7);
// lanes 16-31 -> M=lane-16, K = k0+8..15 and k0+24..31.
static __device__ __forceinline__ v16bf load_a(const __bf16* A, int lda,
                                               int row0, int k0, int lane) {
  int m  = row0 + (lane & 15);
  int kb = k0 + ((lane & 16) >> 1);           // +8 for upper half-wave
  const __bf16* p = A + (size_t)m * lda + kb;
  v8bf lo = *reinterpret_cast<const v8bf*>(p);
  v8bf hi = *reinterpret_cast<const v8bf*>(p + 16);
  v16bf a;
#pragma unroll
  for (int i = 0; i < 8; i++) { a[i] = lo[i]; a[i + 8] = hi[i]; }
  return a;
}

// B-fragment: 32x16 bf16 = W.T tile, W row-major [N, ldw] (y = x @ W.T).
// lane -> N = n0 + lane%16; same K layout as load_a (k0+8h+0..7, k0+16+8h+0..7).
static __device__ __forceinline__ v16bf load_b(const __bf16* __restrict__ W, int ldw,
                                               int n0, int k0, int lane) {
  int n  = n0 + (lane & 15);
  int kb = k0 + ((lane & 16) >> 1);           // +8 for upper half-wave
  const __bf16* p = W + (size_t)n * ldw + kb;
  v8bf lo = *reinterpret_cast<const v8bf*>(p);
  v8bf hi = *reinterpret_cast<const v8bf*>(p + 16);
  v16bf b;
#pragma unroll
  for (int i = 0; i < 8; i++) { b[i] = lo[i]; b[i + 8] = hi[i]; }
  return b;
}

static __device__ __forceinline__ v8f wmma_bf(v16bf a, v16bf b, v8f c) {
  return __builtin_amdgcn_wmma_f32_16x16x32_bf16(false, a, false, b, (short)0, c,
                                                 false, false);
}

// Accumulate two 16x16 output tiles (cols n0a, n0b) over a K range.
// A comes from LDS (staged tile, row0 = 0), B streams from global/L2.
template <int KLEN>
static __device__ __forceinline__ void accum2(v8f& a0, v8f& a1,
                                              const __bf16* A, int lda,
                                              const __bf16* __restrict__ W, int ldw,
                                              int n0a, int n0b, int wk0, int lane) {
#pragma unroll
  for (int kk = 0; kk < KLEN; kk += 32) {
    if (kk + 32 < KLEN)   // warm next B k-block (global_prefetch_b8)
      __builtin_prefetch(W + (size_t)(n0a + (lane & 15)) * ldw + wk0 + kk + 32, 0, 1);
    v16bf af = load_a(A, lda, 0, kk, lane);
    v16bf b0 = load_b(W, ldw, n0a, wk0 + kk, lane);
    a0 = wmma_bf(af, b0, a0);
    v16bf b1 = load_b(W, ldw, n0b, wk0 + kk, lane);
    a1 = wmma_bf(af, b1, a1);
  }
}

// ---------- fused GEMM (+ optional 2nd input segment + rank-2 action term)
// ---------- + bias + LayerNorm + SiLU -> bf16 activations
template <int K1, int K2>
__global__ __launch_bounds__(256) void k_gemm_ln_silu(
    const __bf16* __restrict__ A1, int lda1,
    const __bf16* __restrict__ A2, int lda2,
    const __bf16* __restrict__ W, int ldw,
    const float* __restrict__ act,     // [B,2] slice or null
    const float* __restrict__ WeF,     // f32 We (ld 258), action cols 256..257
    const float* __restrict__ bias, const float* __restrict__ gamma,
    const float* __restrict__ beta, __bf16* __restrict__ out_bf) {
  __shared__ float sm[16 * 272];
  __shared__ alignas(32) __bf16 smA1[16 * K1];
  __shared__ alignas(32) __bf16 smA2[(K2 > 0) ? 16 * K2 : 16];
  const int tid = threadIdx.x, lane = tid & 31, wave = tid >> 5;
  const int row0 = blockIdx.x * 16;
  const int n0a = wave * 32, n0b = n0a + 16;

  stage_tile<K1>(smA1, A1, lda1, row0, tid);
  if constexpr (K2 > 0) stage_tile<K2>(smA2, A2, lda2, row0, tid);
  async_wait_barrier();

  v8f a0{}, a1{};
  accum2<K1>(a0, a1, smA1, K1, W, ldw, n0a, n0b, 0, lane);
  if constexpr (K2 > 0)
    accum2<K2>(a0, a1, smA2, K2, W, ldw, n0a, n0b, K1, lane);

  {
    int rbase = (lane >> 4) << 3;
    int c0 = n0a + (lane & 15), c1 = n0b + (lane & 15);
    float w00 = 0.f, w01 = 0.f, w10 = 0.f, w11 = 0.f;
    if (act) {
      w00 = WeF[c0 * 258 + 256]; w01 = WeF[c0 * 258 + 257];
      w10 = WeF[c1 * 258 + 256]; w11 = WeF[c1 * 258 + 257];
    }
#pragma unroll
    for (int r = 0; r < 8; r++) {
      int rr = rbase + r, m = row0 + rr;
      float v0 = a0[r] + bias[c0];
      float v1 = a1[r] + bias[c1];
      if (act) {
        float x0 = act[m * 2], x1 = act[m * 2 + 1];
        v0 += x0 * w00 + x1 * w01;
        v1 += x0 * w10 + x1 * w11;
      }
      sm[rr * 272 + c0] = v0;
      sm[rr * 272 + c1] = v1;
    }
  }
  __syncthreads();

  // LayerNorm over the 256 columns of each row (16 lanes per row, wave32-safe
  // xor masks 1/2/4/8), then SiLU.
  int row = tid >> 4, cl = tid & 15;
  float s = 0.f, s2 = 0.f;
#pragma unroll
  for (int i = 0; i < 16; i++) {
    float v = sm[row * 272 + cl + 16 * i];
    s += v; s2 += v * v;
  }
#pragma unroll
  for (int off = 8; off; off >>= 1) {
    s  += __shfl_xor(s, off, 32);
    s2 += __shfl_xor(s2, off, 32);
  }
  float mean = s * (1.f / 256.f);
  float var  = s2 * (1.f / 256.f) - mean * mean;
  float inv  = rsqrtf(var + 1e-5f);
  int m = row0 + row;
#pragma unroll
  for (int i = 0; i < 16; i++) {
    int c = cl + 16 * i;
    float v  = (sm[row * 272 + c] - mean) * inv * gamma[c] + beta[c];
    float si = v / (1.f + __expf(-v));
    out_bf[(size_t)m * 256 + c] = f2bf(si);
  }
}

// ---------- fused GRU: 6 WMMA tile accumulations + gating ----------
__global__ __launch_bounds__(256) void k_gru(
    const __bf16* __restrict__ x_bf, const __bf16* __restrict__ h_bf,
    const float* __restrict__ h_f32,
    const __bf16* __restrict__ Wih, const __bf16* __restrict__ Whh,
    const float* __restrict__ bih, const float* __restrict__ bhh,
    float* __restrict__ h_f32_out, __bf16* __restrict__ h_bf_out,
    float* __restrict__ beliefs_out) {
  __shared__ alignas(32) __bf16 smx[16 * 256];
  __shared__ alignas(32) __bf16 smh[16 * 256];
  const int tid = threadIdx.x, lane = tid & 31, wave = tid >> 5;
  const int row0 = blockIdx.x * 16;
  const int n = blockIdx.y * 128 + wave * 16;   // belief column tile

  stage_tile<256>(smx, x_bf, 256, row0, tid);
  stage_tile<256>(smh, h_bf, 256, row0, tid);
  async_wait_barrier();

  v8f air{}, aiz{}, ain{}, ahr{}, ahz{}, ahn{};
#pragma unroll
  for (int kk = 0; kk < 256; kk += 32) {
    v16bf ax = load_a(smx, 256, 0, kk, lane);
    v16bf ah = load_a(smh, 256, 0, kk, lane);
    v16bf b;
    b = load_b(Wih, 256, n,       kk, lane); air = wmma_bf(ax, b, air);
    b = load_b(Wih, 256, n + 256, kk, lane); aiz = wmma_bf(ax, b, aiz);
    b = load_b(Wih, 256, n + 512, kk, lane); ain = wmma_bf(ax, b, ain);
    b = load_b(Whh, 256, n,       kk, lane); ahr = wmma_bf(ah, b, ahr);
    b = load_b(Whh, 256, n + 256, kk, lane); ahz = wmma_bf(ah, b, ahz);
    b = load_b(Whh, 256, n + 512, kk, lane); ahn = wmma_bf(ah, b, ahn);
  }
  int rbase = (lane >> 4) << 3;
  int c = n + (lane & 15);
  float bir = bih[c], biz = bih[c + 256], bin_ = bih[c + 512];
  float bhr = bhh[c], bhz = bhh[c + 256], bhn  = bhh[c + 512];
#pragma unroll
  for (int r = 0; r < 8; r++) {
    int m = row0 + rbase + r;
    float ir = air[r] + bir, iz = aiz[r] + biz, inn = ain[r] + bin_;
    float hr = ahr[r] + bhr, hz = ahz[r] + bhz, hn  = ahn[r] + bhn;
    float rg = 1.f / (1.f + __expf(-(ir + hr)));
    float z  = 1.f / (1.f + __expf(-(iz + hz)));
    float nn = tanhf(inn + rg * hn);
    float hp = h_f32[(size_t)m * 256 + c];
    float hv = (1.f - z) * nn + z * hp;
    size_t idx = (size_t)m * 256 + c;
    h_f32_out[idx]   = hv;
    h_bf_out[idx]    = f2bf(hv);
    beliefs_out[idx] = hv;
  }
}

// ---------- fused logits GEMM + unimix softmax + Gumbel-max one-hot ----------
__global__ __launch_bounds__(256) void k_logits_sample(
    const __bf16* __restrict__ Hin, const __bf16* __restrict__ W,
    const float* __restrict__ bias,
    float* __restrict__ logits_out, float* __restrict__ state_out,
    __bf16* __restrict__ state_bf, uint32_t salt, int t) {
  __shared__ float sm[16 * 272];
  __shared__ alignas(32) __bf16 smA[16 * 256];
  const int tid = threadIdx.x, lane = tid & 31, wave = tid >> 5;
  const int row0 = blockIdx.x * 16;
  const int n0a = wave * 32, n0b = n0a + 16;

  stage_tile<256>(smA, Hin, 256, row0, tid);
  async_wait_barrier();

  v8f a0{}, a1{};
  accum2<256>(a0, a1, smA, 256, W, 256, n0a, n0b, 0, lane);

  {
    int rbase = (lane >> 4) << 3;
    int c0 = n0a + (lane & 15), c1 = n0b + (lane & 15);
#pragma unroll
    for (int r = 0; r < 8; r++) {
      int rr = rbase + r, m = row0 + rr;
      float v0 = a0[r] + bias[c0];
      float v1 = a1[r] + bias[c1];
      sm[rr * 272 + c0] = v0;
      sm[rr * 272 + c1] = v1;
      logits_out[(size_t)m * 256 + c0] = v0;
      logits_out[(size_t)m * 256 + c1] = v1;
    }
  }
  __syncthreads();

  // one thread per (row, category): softmax(16) -> unimix -> log ->
  // softmax -> Gumbel-max -> one-hot (the ST-grad term cancels in forward).
  int row = tid >> 4, cat = tid & 15;
  int m = row0 + row;
  float v[16];
  float mx = -1e30f;
#pragma unroll
  for (int j = 0; j < 16; j++) {
    v[j] = sm[row * 272 + cat * 16 + j];
    mx = fmaxf(mx, v[j]);
  }
  float se = 0.f;
#pragma unroll
  for (int j = 0; j < 16; j++) { v[j] = __expf(v[j] - mx); se += v[j]; }
  float invs = 1.f / se;
  float mx2 = -1e30f;
#pragma unroll
  for (int j = 0; j < 16; j++) {
    float p = 0.99f * v[j] * invs + 0.01f / 16.f;
    v[j] = __logf(p + 1e-8f);
    mx2 = fmaxf(mx2, v[j]);
  }
  float se2 = 0.f;
#pragma unroll
  for (int j = 0; j < 16; j++) se2 += __expf(v[j] - mx2);
  float lse2 = __logf(se2);
  uint32_t base = hashu(salt * 0x9E3779B9u ^ (uint32_t)(t * 0x85EBCA6Bu) ^ 0xC0FFEEu);
  int best = 0; float bs = -1e30f;
#pragma unroll
  for (int j = 0; j < 16; j++) {
    float logpr = v[j] - mx2 - lse2;
    float sc = logpr + gumbel_from(hashu(base ^ (uint32_t)((m << 8) | (cat << 4) | j)));
    if (sc > bs) { bs = sc; best = j; }
  }
#pragma unroll
  for (int j = 0; j < 16; j++) {
    float o = (j == best) ? 1.f : 0.f;
    size_t idx = (size_t)m * 256 + cat * 16 + j;
    state_out[idx] = o;
    if (state_bf) state_bf[idx] = f2bf(o);
  }
}

// ---------- helpers ----------
__global__ void k_cvt2d(const float* __restrict__ src, __bf16* __restrict__ dst,
                        int rows, int cols, int src_ld) {
  int i = blockIdx.x * blockDim.x + threadIdx.x;
  if (i >= rows * cols) return;
  int r = i / cols, c = i % cols;
  dst[i] = f2bf(src[(size_t)r * src_ld + c]);
}
__global__ void k_copy_f32(const float* __restrict__ s, float* __restrict__ d, int n) {
  int i = blockIdx.x * blockDim.x + threadIdx.x;
  if (i < n) d[i] = s[i];
}

extern "C" void kernel_launch(void* const* d_in, const int* in_sizes, int n_in,
                              void* d_out, int out_size, void* d_ws, size_t ws_size,
                              hipStream_t stream) {
  (void)in_sizes; (void)n_in; (void)out_size; (void)ws_size;
  const float* prev_state   = (const float*)d_in[0];
  const float* actions      = (const float*)d_in[1];
  const float* prev_belief  = (const float*)d_in[2];
  const float* observations = (const float*)d_in[3];
  const float* We  = (const float*)d_in[4];
  const float* be  = (const float*)d_in[5];
  const float* ge  = (const float*)d_in[6];
  const float* bte = (const float*)d_in[7];
  const float* Wih = (const float*)d_in[8];
  const float* Whh = (const float*)d_in[9];
  const float* bih = (const float*)d_in[10];
  const float* bhh = (const float*)d_in[11];
  const float* Wp1 = (const float*)d_in[12];
  const float* bp1 = (const float*)d_in[13];
  const float* gp  = (const float*)d_in[14];
  const float* btp = (const float*)d_in[15];
  const float* Wp2 = (const float*)d_in[16];
  const float* bp2 = (const float*)d_in[17];
  const float* Wq1 = (const float*)d_in[18];
  const float* bq1 = (const float*)d_in[19];
  const float* gq  = (const float*)d_in[20];
  const float* btq = (const float*)d_in[21];
  const float* Wq2 = (const float*)d_in[22];
  const float* bq2 = (const float*)d_in[23];

  // workspace carve-out (~7 MB, 256B aligned)
  char* wsp = (char*)d_ws;
  auto alloc = [&](size_t bytes) {
    char* p = wsp;
    wsp += (bytes + 255) & ~(size_t)255;
    return p;
  };
  __bf16* We_bf  = (__bf16*)alloc((size_t)256 * 256 * 2);
  __bf16* Wih_bf = (__bf16*)alloc((size_t)768 * 256 * 2);
  __bf16* Whh_bf = (__bf16*)alloc((size_t)768 * 256 * 2);
  __bf16* Wp1_bf = (__bf16*)alloc((size_t)256 * 256 * 2);
  __bf16* Wp2_bf = (__bf16*)alloc((size_t)256 * 256 * 2);
  __bf16* Wq1_bf = (__bf16*)alloc((size_t)256 * 768 * 2);
  __bf16* Wq2_bf = (__bf16*)alloc((size_t)256 * 256 * 2);
  __bf16* state_bf = (__bf16*)alloc((size_t)Bb * 256 * 2);
  __bf16* belief_bf[2] = { (__bf16*)alloc((size_t)Bb * 256 * 2),
                           (__bf16*)alloc((size_t)Bb * 256 * 2) };
  float*  belief_f32[2] = { (float*)alloc((size_t)Bb * 256 * 4),
                            (float*)alloc((size_t)Bb * 256 * 4) };
  __bf16* x_bf  = (__bf16*)alloc((size_t)Bb * 256 * 2);
  __bf16* h1_bf = (__bf16*)alloc((size_t)Bb * 256 * 2);
  __bf16* h2_bf = (__bf16*)alloc((size_t)Bb * 256 * 2);
  __bf16* obs_bf = (__bf16*)alloc((size_t)Bb * 512 * 2);

  auto cvt = [&](const float* src, __bf16* dst, int rows, int cols, int ld) {
    int n = rows * cols;
    k_cvt2d<<<(n + 255) / 256, 256, 0, stream>>>(src, dst, rows, cols, ld);
  };
  cvt(We,  We_bf,  256, 256, 258);   // drop the 2 action cols (handled in f32)
  cvt(Wih, Wih_bf, 768, 256, 256);
  cvt(Whh, Whh_bf, 768, 256, 256);
  cvt(Wp1, Wp1_bf, 256, 256, 256);
  cvt(Wp2, Wp2_bf, 256, 256, 256);
  cvt(Wq1, Wq1_bf, 256, 768, 768);
  cvt(Wq2, Wq2_bf, 256, 256, 256);
  cvt(prev_state,  state_bf,     Bb, 256, 256);
  cvt(prev_belief, belief_bf[0], Bb, 256, 256);
  k_copy_f32<<<(Bb * 256 + 255) / 256, 256, 0, stream>>>(prev_belief,
                                                         belief_f32[0], Bb * 256);

  float* out = (float*)d_out;
  const size_t NTB = (size_t)Tt * Bb * 256;
  float* out_beliefs = out;
  float* out_plog    = out + NTB;
  float* out_pstate  = out + 2 * NTB;
  float* out_qlog    = out + 3 * NTB;
  float* out_qstate  = out + 4 * NTB;

  for (int t = 0; t < Tt; t++) {
    int pin = t & 1, pout = 1 - pin;
    const size_t tofs = (size_t)t * Bb * 256;
    const float* act_t = actions + (size_t)t * Bb * 2;
    const float* obs_t = observations + (size_t)t * Bb * 512;

    // obs -> bf16 staging for the posterior GEMM
    k_cvt2d<<<(Bb * 512 + 255) / 256, 256, 0, stream>>>(obs_t, obs_bf, Bb, 512, 512);

    // x = silu(LN([state, a] @ We.T + be))
    k_gemm_ln_silu<256, 0><<<Bb / 16, 256, 0, stream>>>(
        state_bf, 256, nullptr, 0, We_bf, 256, act_t, We, be, ge, bte, x_bf);

    // belief = GRU(x, belief)
    k_gru<<<dim3(Bb / 16, 2), 256, 0, stream>>>(
        x_bf, belief_bf[pin], belief_f32[pin], Wih_bf, Whh_bf, bih, bhh,
        belief_f32[pout], belief_bf[pout], out_beliefs + tofs);

    // h1 = silu(LN(belief @ Wp1.T + bp1)); prior logits + sample
    k_gemm_ln_silu<256, 0><<<Bb / 16, 256, 0, stream>>>(
        belief_bf[pout], 256, nullptr, 0, Wp1_bf, 256,
        nullptr, nullptr, bp1, gp, btp, h1_bf);
    k_logits_sample<<<Bb / 16, 256, 0, stream>>>(
        h1_bf, Wp2_bf, bp2, out_plog + tofs, out_pstate + tofs, nullptr, 0u, t);

    // h2 = silu(LN([belief, o] @ Wq1.T + bq1)); posterior logits + sample
    k_gemm_ln_silu<256, 512><<<Bb / 16, 256, 0, stream>>>(
        belief_bf[pout], 256, obs_bf, 512, Wq1_bf, 768,
        nullptr, nullptr, bq1, gq, btq, h2_bf);
    k_logits_sample<<<Bb / 16, 256, 0, stream>>>(
        h2_bf, Wq2_bf, bq2, out_qlog + tofs, out_qstate + tofs, state_bf, 1u, t);
  }
}
